// THALNET_77910706749834
// MI455X (gfx1250) — matrix operand order repaired and checked by reading, not compiled, not run on hardware
//
#include <hip/hip_runtime.h>
#include <cstdint>
#include <cstddef>

typedef unsigned short u16;
typedef __attribute__((ext_vector_type(16))) __bf16 v16bf;
typedef __attribute__((ext_vector_type(8)))  float  v8f;

#define WMMA_BF16(A_, B_, C_) \
  __builtin_amdgcn_wmma_f32_16x16x32_bf16(false, (A_), false, (B_), (short)0, (C_), false, false)

// ---------------------------------------------------------------------------
// helpers
// ---------------------------------------------------------------------------
__device__ __forceinline__ u16 f2bf(float f) {
  union { float f; unsigned u; } c; c.f = f;
  unsigned u = c.u + 0x7FFFu + ((c.u >> 16) & 1u);   // round-to-nearest-even
  return (u16)(u >> 16);
}

// A fragment 16x32 bf16 (ISA 7.12.2): lanes 0-15 -> rows, K {0..7,16..23};
// lanes 16-31 -> rows, K {8..15,24..31}. LDS tile row stride = 40 elements.
__device__ __forceinline__ v16bf load_a_frag(const u16* base, int lane) {
  const int half = lane >> 4;
  const u16* p = base + (lane & 15) * 40 + half * 8;
  union { uint4 u[2]; v16bf v; } c;
  c.u[0] = *reinterpret_cast<const uint4*>(p);
  c.u[1] = *reinterpret_cast<const uint4*>(p + 16);
  return c.v;
}

// B fragment 32x16 bf16: lane n<16 holds col n, K {0..7,16..23}; lanes 16-31 K {8..15,24..31}.
// LDS tile stored N-major (weights are (N,K) row-major), stride 40.
__device__ __forceinline__ v16bf load_b_frag(const u16* base, int lane) {
  const int half = lane >> 4;
  const u16* p = base + (lane & 15) * 40 + half * 8;
  union { uint4 u[2]; v16bf v; } c;
  c.u[0] = *reinterpret_cast<const uint4*>(p);
  c.u[1] = *reinterpret_cast<const uint4*>(p + 16);
  return c.v;
}

// ---------------------------------------------------------------------------
// prep kernels
// ---------------------------------------------------------------------------
__global__ void thal_conv_x(const float* __restrict__ in, u16* __restrict__ outb) {
  int idx = blockIdx.x * 256 + threadIdx.x;       // (s,b,i), 256*128*128 exact
  int s = idx >> 14, rem = idx & 16383;
  int b = rem >> 7, i = rem & 127;
  outb[idx] = f2bf(in[(size_t)b * 32768 + s * 128 + i]);
}

__global__ void thal_pm_tr(const float* __restrict__ p0, const float* __restrict__ p1,
                           const float* __restrict__ p2, const float* __restrict__ p3,
                           u16* __restrict__ dst) {
  int idx = blockIdx.x * 256 + threadIdx.x;       // (n,k), 1024*4096 exact
  int n = idx >> 12, k = idx & 4095;
  const float* pm = (n < 256) ? p0 : (n < 512) ? p1 : (n < 768) ? p2 : p3;
  dst[idx] = f2bf(pm[(size_t)k * 256 + (n & 255)]);
}

__global__ void thal_conv(const float* __restrict__ src, u16* __restrict__ dst, int n) {
  int idx = blockIdx.x * 256 + threadIdx.x;
  if (idx < n) dst[idx] = f2bf(src[idx]);
}

__global__ void thal_init_center(const float* __restrict__ ci, u16* __restrict__ cen) {
  int idx = blockIdx.x * 256 + threadIdx.x;       // 128*4096 exact
  int b = idx >> 12, col = idx & 4095;
  int im = col >> 10, c = col & 1023;
  cen[idx] = f2bf(ci[(size_t)im * 131072 + b * 1024 + c]);
}

__global__ void thal_init_h(const float* __restrict__ mi, const float* __restrict__ ml,
                            float* __restrict__ hF, u16* __restrict__ hB) {
  int idx = blockIdx.x * 256 + threadIdx.x;       // 540672 exact (2112*256)
  float v;
  if (idx < 393216) v = mi[idx];                  // modules 0..2, layout matches
  else              v = ml[idx - 393216];         // module 3 (B,1152)
  hF[idx] = v;
  hB[idx] = f2bf(v);
}

// ---------------------------------------------------------------------------
// Kernel A: ctx_all(128,1024) = center(128,4096) @ pmT(1024,4096)^T   (bf16 out)
// grid = 16 blocks of 256 threads; block tile 128x64; wave tile 32x32.
// ---------------------------------------------------------------------------
__global__ __launch_bounds__(256) void thal_ctx_gemm(
    const u16* __restrict__ A, const u16* __restrict__ W, u16* __restrict__ C) {
  __shared__ __align__(16) u16 sA[128 * 40];
  __shared__ __align__(16) u16 sB[64 * 40];
  const int tid = threadIdx.x;
  const int lane = tid & 31, wave = tid >> 5;
  const int wr = (wave & 3) * 32, wc = (wave >> 2) * 32;
  const int n0 = blockIdx.x * 64;
  v8f acc[2][2] = {};

  for (int k0 = 0; k0 < 4096; k0 += 32) {
#pragma unroll
    for (int c = 0; c < 2; ++c) {
      const int chunk = tid * 2 + c;
      const int row = chunk >> 2, part = chunk & 3;
      *reinterpret_cast<uint4*>(&sA[row * 40 + part * 8]) =
          *reinterpret_cast<const uint4*>(A + (size_t)row * 4096 + k0 + part * 8);
    }
    {
      const int row = tid >> 2, part = tid & 3;
      *reinterpret_cast<uint4*>(&sB[row * 40 + part * 8]) =
          *reinterpret_cast<const uint4*>(W + (size_t)(n0 + row) * 4096 + k0 + part * 8);
    }
    __syncthreads();
    v16bf a0 = load_a_frag(sA + (wr + 0) * 40, lane);
    v16bf a1 = load_a_frag(sA + (wr + 16) * 40, lane);
    v16bf b0 = load_b_frag(sB + (wc + 0) * 40, lane);
    v16bf b1 = load_b_frag(sB + (wc + 16) * 40, lane);
    acc[0][0] = WMMA_BF16(a0, b0, acc[0][0]);
    acc[0][1] = WMMA_BF16(a0, b1, acc[0][1]);
    acc[1][0] = WMMA_BF16(a1, b0, acc[1][0]);
    acc[1][1] = WMMA_BF16(a1, b1, acc[1][1]);
    __syncthreads();
  }
#pragma unroll
  for (int mi = 0; mi < 2; ++mi)
#pragma unroll
    for (int ni = 0; ni < 2; ++ni) {
      const int n = n0 + wc + ni * 16 + (lane & 15);
#pragma unroll
      for (int v = 0; v < 8; ++v) {
        const int m = wr + mi * 16 + ((lane < 16) ? v : v + 8);
        C[(size_t)m * 1024 + n] = f2bf(acc[mi][ni][v]);
      }
    }
}

// ---------------------------------------------------------------------------
// Kernel B: fused GRU step for all 4 modules. grid = 66 blocks:
//   blocks [0,16) mod0, [16,32) mod1, [32,48) mod2, [48,66) mod3 (H=1152).
// Block computes a 128x64 slice of h'. r/z accumulate gi+gh together;
// n-gate keeps inn (aN) and hn (aHN) separate. Epilogue = gate math +
// state/center/output writeback.
// ---------------------------------------------------------------------------
__global__ __launch_bounds__(256) void thal_gru_step(
    int t,
    const u16* __restrict__ xbf, const u16* __restrict__ ctx,
    const float* __restrict__ hInF, const u16* __restrict__ hInB,
    float* __restrict__ hOutF, u16* __restrict__ hOutB,
    u16* __restrict__ centerOut,
    const u16* __restrict__ wihB, const u16* __restrict__ whhB,
    const float* __restrict__ bih0, const float* __restrict__ bih1,
    const float* __restrict__ bih2, const float* __restrict__ bih3,
    const float* __restrict__ bhh0, const float* __restrict__ bhh1,
    const float* __restrict__ bhh2, const float* __restrict__ bhh3,
    float* __restrict__ out) {
  __shared__ __align__(16) u16 sA[128 * 40];
  __shared__ __align__(16) u16 sB[3 * 64 * 40];

  const int tid = threadIdx.x;
  const int lane = tid & 31, wave = tid >> 5;
  const int wr = (wave & 3) * 32, wc = (wave >> 2) * 32;

  const int bx = blockIdx.x;
  int mod, nt;
  if (bx < 16)      { mod = 0; nt = bx; }
  else if (bx < 32) { mod = 1; nt = bx - 16; }
  else if (bx < 48) { mod = 2; nt = bx - 32; }
  else              { mod = 3; nt = bx - 48; }

  const int H   = (mod == 3) ? 1152 : 1024;
  const int KIN = (mod == 0) ? 384 : 256;
  const int wihOffA[4] = {0, 1179648, 1966080, 2752512};
  const int whhOffA[4] = {0, 3145728, 6291456, 9437184};
  const int hOffA[4]   = {0, 131072, 262144, 393216};
  const int wihO = wihOffA[mod], whhO = whhOffA[mod], hO = hOffA[mod];
  const int n0 = nt * 64;

  v8f aR[2][2] = {}, aZ[2][2] = {}, aN[2][2] = {}, aHN[2][2] = {};

  auto mmaGate = [&](v8f (&acc)[2][2], const v16bf& a0, const v16bf& a1, const u16* sBg) {
    v16bf b0 = load_b_frag(sBg + (wc + 0) * 40, lane);
    v16bf b1 = load_b_frag(sBg + (wc + 16) * 40, lane);
    acc[0][0] = WMMA_BF16(a0, b0, acc[0][0]);
    acc[0][1] = WMMA_BF16(a0, b1, acc[0][1]);
    acc[1][0] = WMMA_BF16(a1, b0, acc[1][0]);
    acc[1][1] = WMMA_BF16(a1, b1, acc[1][1]);
  };

  // ---- phase 1: input path (xin @ wih^T) ----
  for (int k0 = 0; k0 < KIN; k0 += 32) {
#pragma unroll
    for (int c = 0; c < 2; ++c) {
      const int chunk = tid * 2 + c;
      const int row = chunk >> 2, part = chunk & 3;
      const int gcol = k0 + part * 8;
      const u16* src;
      if (mod == 0) {
        src = (gcol < 128) ? (xbf + ((size_t)t * 128 + row) * 128 + gcol)
                           : (ctx + (size_t)row * 1024 + (gcol - 128));
      } else {
        src = ctx + (size_t)row * 1024 + mod * 256 + gcol;
      }
      *reinterpret_cast<uint4*>(&sA[row * 40 + part * 8]) =
          *reinterpret_cast<const uint4*>(src);
    }
#pragma unroll
    for (int g = 0; g < 3; ++g) {
      const int row = tid >> 2, part = tid & 3;
      const u16* src = wihB + wihO + (size_t)(g * H + n0 + row) * KIN + k0 + part * 8;
      *reinterpret_cast<uint4*>(&sB[g * 2560 + row * 40 + part * 8]) =
          *reinterpret_cast<const uint4*>(src);
    }
    __syncthreads();
    v16bf a0 = load_a_frag(sA + (wr + 0) * 40, lane);
    v16bf a1 = load_a_frag(sA + (wr + 16) * 40, lane);
    mmaGate(aR, a0, a1, sB + 0);
    mmaGate(aZ, a0, a1, sB + 2560);
    mmaGate(aN, a0, a1, sB + 5120);
    __syncthreads();
  }

  // ---- phase 2: hidden path (h @ whh^T) ----
  for (int k0 = 0; k0 < H; k0 += 32) {
#pragma unroll
    for (int c = 0; c < 2; ++c) {
      const int chunk = tid * 2 + c;
      const int row = chunk >> 2, part = chunk & 3;
      const u16* src = hInB + hO + (size_t)row * H + k0 + part * 8;
      *reinterpret_cast<uint4*>(&sA[row * 40 + part * 8]) =
          *reinterpret_cast<const uint4*>(src);
    }
#pragma unroll
    for (int g = 0; g < 3; ++g) {
      const int row = tid >> 2, part = tid & 3;
      const u16* src = whhB + whhO + (size_t)(g * H + n0 + row) * H + k0 + part * 8;
      *reinterpret_cast<uint4*>(&sB[g * 2560 + row * 40 + part * 8]) =
          *reinterpret_cast<const uint4*>(src);
    }
    __syncthreads();
    v16bf a0 = load_a_frag(sA + (wr + 0) * 40, lane);
    v16bf a1 = load_a_frag(sA + (wr + 16) * 40, lane);
    mmaGate(aR, a0, a1, sB + 0);
    mmaGate(aZ, a0, a1, sB + 2560);
    mmaGate(aHN, a0, a1, sB + 5120);
    __syncthreads();
  }

  // ---- epilogue: GRU gate math + writeback ----
  const float* bih = (mod == 0) ? bih0 : (mod == 1) ? bih1 : (mod == 2) ? bih2 : bih3;
  const float* bhh = (mod == 0) ? bhh0 : (mod == 1) ? bhh1 : (mod == 2) ? bhh2 : bhh3;

#pragma unroll
  for (int mi = 0; mi < 2; ++mi) {
#pragma unroll
    for (int ni = 0; ni < 2; ++ni) {
      const int ncol = n0 + wc + ni * 16 + (lane & 15);
      const float br  = bih[ncol] + bhh[ncol];
      const float bz  = bih[H + ncol] + bhh[H + ncol];
      const float bni = bih[2 * H + ncol];
      const float bnh = bhh[2 * H + ncol];
#pragma unroll
      for (int v = 0; v < 8; ++v) {
        const int m = wr + mi * 16 + ((lane < 16) ? v : v + 8);
        const float hp = hInF[hO + (size_t)m * H + ncol];
        const float r  = 1.0f / (1.0f + __expf(-(aR[mi][ni][v] + br)));
        const float z  = 1.0f / (1.0f + __expf(-(aZ[mi][ni][v] + bz)));
        const float nn = tanhf(aN[mi][ni][v] + bni + r * (aHN[mi][ni][v] + bnh));
        const float hnew = (1.0f - z) * nn + z * hp;
        hOutF[hO + (size_t)m * H + ncol] = hnew;
        const u16 hb = f2bf(hnew);
        hOutB[hO + (size_t)m * H + ncol] = hb;
        if (mod < 3) {
          centerOut[(size_t)m * 4096 + mod * 1024 + ncol] = hb;
        } else if (ncol >= 128) {
          centerOut[(size_t)m * 4096 + 3072 + (ncol - 128)] = hb;
        } else {
          out[(size_t)m * 32768 + t * 128 + ncol] = hnew;   // (B,S,OUT)
        }
      }
    }
  }
}

// ---------------------------------------------------------------------------
// host driver
// ---------------------------------------------------------------------------
extern "C" void kernel_launch(void* const* d_in, const int* in_sizes, int n_in,
                              void* d_out, int out_size, void* d_ws, size_t ws_size,
                              hipStream_t stream) {
  (void)in_sizes; (void)n_in; (void)out_size;
  const float* inputs = (const float*)d_in[0];
  const float* cinit  = (const float*)d_in[1];
  const float* minit  = (const float*)d_in[2];
  const float* minitl = (const float*)d_in[3];
  const float *pm[4], *wih[4], *whh[4], *bih[4], *bhh[4];
  for (int i = 0; i < 4; ++i) {
    pm[i]  = (const float*)d_in[4 + 5 * i];
    wih[i] = (const float*)d_in[5 + 5 * i];
    whh[i] = (const float*)d_in[6 + 5 * i];
    bih[i] = (const float*)d_in[7 + 5 * i];
    bhh[i] = (const float*)d_in[8 + 5 * i];
  }
  float* out = (float*)d_out;

  char* ws = (char*)d_ws;
  size_t off = 0;
  auto take = [&](size_t bytes) -> char* {
    char* p = ws + off;
    off += (bytes + 255) & ~(size_t)255;
    return p;
  };
  u16* xbf    = (u16*)take(8388608);            // (S,B,I) bf16
  u16* pmT    = (u16*)take(8388608);            // (1024,4096) bf16
  u16* wihBf  = (u16*)take(7274496);            // packed bf16
  u16* whhBf  = (u16*)take(26836992);           // packed bf16
  u16* ctx    = (u16*)take(262144);             // (128,1024) bf16
  u16* cen[2]  = {(u16*)take(1048576), (u16*)take(1048576)};
  float* hF[2] = {(float*)take(2162688), (float*)take(2162688)};
  u16* hB[2]   = {(u16*)take(1081344), (u16*)take(1081344)};
  if (off > ws_size) return;

  thal_conv_x<<<16384, 256, 0, stream>>>(inputs, xbf);
  thal_pm_tr<<<16384, 256, 0, stream>>>(pm[0], pm[1], pm[2], pm[3], pmT);

  const int wihSz[4] = {1179648, 786432, 786432, 884736};
  const int wihOf[4] = {0, 1179648, 1966080, 2752512};
  const int whhSz[4] = {3145728, 3145728, 3145728, 3981312};
  const int whhOf[4] = {0, 3145728, 6291456, 9437184};
  for (int i = 0; i < 4; ++i) {
    thal_conv<<<(wihSz[i] + 255) / 256, 256, 0, stream>>>(wih[i], wihBf + wihOf[i], wihSz[i]);
    thal_conv<<<(whhSz[i] + 255) / 256, 256, 0, stream>>>(whh[i], whhBf + whhOf[i], whhSz[i]);
  }
  thal_init_center<<<2048, 256, 0, stream>>>(cinit, cen[0]);
  thal_init_h<<<2112, 256, 0, stream>>>(minit, minitl, hF[0], hB[0]);

  for (int t = 0; t < 256; ++t) {
    const int p = t & 1;
    thal_ctx_gemm<<<16, 256, 0, stream>>>(cen[p], pmT, ctx);
    thal_gru_step<<<66, 256, 0, stream>>>(
        t, xbf, ctx, hF[p], hB[p], hF[1 - p], hB[1 - p], cen[1 - p],
        wihBf, whhBf,
        bih[0], bih[1], bih[2], bih[3],
        bhh[0], bhh[1], bhh[2], bhh[3], out);
  }
}
